// GNNClassifier_88648124990608
// MI455X (gfx1250) — matrix operand order repaired and checked by reading, not compiled, not run on hardware
//
#include <hip/hip_runtime.h>

typedef __attribute__((ext_vector_type(16))) _Float16 v16h;
typedef __attribute__((ext_vector_type(8)))  float    v8f;

#define HID 64

// ---------------------------------------------------------------- utilities

__global__ void zero_kernel(float* __restrict__ p, long long n) {
    long long i = (long long)blockIdx.x * blockDim.x + threadIdx.x;
    if (i < n) p[i] = 0.0f;
}

// atomic histogram: cnt[ids[i]] += 1
__global__ void count_kernel(const int* __restrict__ ids, float* __restrict__ cnt, int n) {
    int i = blockIdx.x * blockDim.x + threadIdx.x;
    if (i < n) atomicAdd(&cnt[ids[i]], 1.0f);
}

// ------------------------------------------------------- embedding lookup

__global__ void embed_kernel(const int* __restrict__ tok, const float* __restrict__ E,
                             float* __restrict__ x, int N) {
    int id = blockIdx.x * blockDim.x + threadIdx.x;
    if (id >= N * 16) return;
    int i = id >> 4, q = id & 15;
    int t = tok[i];
    float4 v = *(const float4*)(E + (size_t)t * HID + q * 4);
    *(float4*)(x + (size_t)i * HID + q * 4) = v;
}

// ------------------------------------------------- edge gather/scatter-add

__global__ void scatter_kernel(const float* __restrict__ x, float* __restrict__ agg,
                               const int* __restrict__ src, const int* __restrict__ dst,
                               int E) {
    int id = blockIdx.x * blockDim.x + threadIdx.x;
    if (id >= E * 16) return;
    int e = id >> 4, q = id & 15;
    int s = src[e], d = dst[e];
    float4 v = *(const float4*)(x + (size_t)s * HID + q * 4);
    float* p = agg + (size_t)d * HID + q * 4;
    atomicAdd(p + 0, v.x);
    atomicAdd(p + 1, v.y);
    atomicAdd(p + 2, v.z);
    atomicAdd(p + 3, v.w);
}

// ------------------------------------------------- weight packing (f32 -> f16 B-layout)
// B-matrix 32x16 f16, 8 VGPRs/lane: lane half kb = lane>>4; VGPR j<4 packs
// K = 2j + 8*kb (+1), VGPR j>=4 packs K = 16 + 2(j-4) + 8*kb (+1), column N = lane&15.
// Packed index layout: P[(((t*2)+c)*32 + lane)*8 + j]  (t = 16-col output tile,
// c = 32-row K chunk) so the hot loop reads two contiguous uint4 per B tile.

__global__ void pack_w_kernel(const float* __restrict__ W, unsigned* __restrict__ P) {
    int idx = blockIdx.x * blockDim.x + threadIdx.x;
    if (idx >= 2048) return;
    int j    = idx & 7;
    int lane = (idx >> 3) & 31;
    int c    = (idx >> 8) & 1;
    int t    = idx >> 9;
    int col  = lane & 15;
    int kb   = lane >> 4;
    int k0   = c * 32 + ((j < 4) ? (2 * j + 8 * kb) : (16 + 2 * (j - 4) + 8 * kb));
    int rowo = t * 16 + col;                 // output feature (row of W)
    union { _Float16 h[2]; unsigned u; } cu;
    cu.h[0] = (_Float16)W[rowo * HID + k0];
    cu.h[1] = (_Float16)W[rowo * HID + k0 + 1];
    P[idx] = cu.u;
}

// ----------------------------------------------------------- SAGE combine
// A-matrix 16x32 f16 per ISA layout: lane half hv = lane>>4, row M = lane&15.
// VGPR j<4:  K = 2j + 8*hv ;  VGPR j>=4: K = 16 + 2(j-4) + 8*hv  (+32 for chunk 1)

__device__ __forceinline__ v16h pack_A(const float* __restrict__ rp, int c, int hv, float rs) {
    v16h a;
#pragma unroll
    for (int j = 0; j < 8; ++j) {
        int k0 = c * 32 + ((j < 4) ? (2 * j + 8 * hv) : (16 + 2 * (j - 4) + 8 * hv));
        float2 f = *(const float2*)(rp + k0);
        a[2 * j]     = (_Float16)(f.x * rs);
        a[2 * j + 1] = (_Float16)(f.y * rs);
    }
    return a;
}

__device__ __forceinline__ v16h load_B(const unsigned* __restrict__ P, int t, int c, int lane) {
    const unsigned* p = P + (((t * 2 + c) * 32 + lane) * 8);
    union { uint4 u[2]; v16h h; } cvt;
    cvt.u[0] = *(const uint4*)p;
    cvt.u[1] = *(const uint4*)(p + 4);
    return cvt.h;
}

// out = relu( (agg/deg) @ Wl^T + bl + x @ Wr^T ), one wave per 16 nodes
__global__ __launch_bounds__(256) void sage_kernel(
    const float* __restrict__ xin, const float* __restrict__ agg,
    const float* __restrict__ deg, const unsigned* __restrict__ Pl,
    const unsigned* __restrict__ Pr, const float* __restrict__ bl,
    float* __restrict__ xout, int N)
{
    int wave = (blockIdx.x * blockDim.x + threadIdx.x) >> 5;
    int lane = threadIdx.x & 31;
    int m0 = wave * 16;
    if (m0 >= N) return;                       // wave-uniform: EXEC stays all-ones
    int M = lane & 15, hv = lane >> 4;
    int row = m0 + M; if (row >= N) row = N - 1;   // clamp tail rows (stores predicated)
    float rs = 1.0f / fmaxf(deg[row], 1.0f);

    v8f acc[4];
#pragma unroll
    for (int t = 0; t < 4; ++t) {
        float b = bl[t * 16 + M];              // bias depends on output column only
#pragma unroll
        for (int r = 0; r < 8; ++r) acc[t][r] = b;
    }

    {   // mean-aggregated neighbor term through W_l
        const float* rp = agg + (size_t)row * HID;
        v16h a0 = pack_A(rp, 0, hv, rs);
        v16h a1 = pack_A(rp, 1, hv, rs);
#pragma unroll
        for (int t = 0; t < 4; ++t) {
            acc[t] = __builtin_amdgcn_wmma_f32_16x16x32_f16(
                false, a0, false, load_B(Pl, t, 0, lane), (short)0, acc[t], false, false);
            acc[t] = __builtin_amdgcn_wmma_f32_16x16x32_f16(
                false, a1, false, load_B(Pl, t, 1, lane), (short)0, acc[t], false, false);
        }
    }
    {   // self term through W_r
        const float* rp = xin + (size_t)row * HID;
        v16h a0 = pack_A(rp, 0, hv, 1.0f);
        v16h a1 = pack_A(rp, 1, hv, 1.0f);
#pragma unroll
        for (int t = 0; t < 4; ++t) {
            acc[t] = __builtin_amdgcn_wmma_f32_16x16x32_f16(
                false, a0, false, load_B(Pr, t, 0, lane), (short)0, acc[t], false, false);
            acc[t] = __builtin_amdgcn_wmma_f32_16x16x32_f16(
                false, a1, false, load_B(Pr, t, 1, lane), (short)0, acc[t], false, false);
        }
    }

    // C/D layout: VGPR r -> node row (r + 8*hv), column N = t*16 + (lane&15)
#pragma unroll
    for (int t = 0; t < 4; ++t) {
#pragma unroll
        for (int r = 0; r < 8; ++r) {
            int node = m0 + r + 8 * hv;
            if (node < N)
                xout[(size_t)node * HID + t * 16 + M] = fmaxf(acc[t][r], 0.0f);
        }
    }
}

// ------------------------------------------------------------ mean pool

__global__ void pool_kernel(const float* __restrict__ x, const int* __restrict__ batch,
                            float* __restrict__ pooled, int N) {
    int id = blockIdx.x * blockDim.x + threadIdx.x;
    if (id >= N * 16) return;
    int i = id >> 4, q = id & 15;
    int g = batch[i];
    float4 v = *(const float4*)(x + (size_t)i * HID + q * 4);
    float* p = pooled + (size_t)g * HID + q * 4;
    atomicAdd(p + 0, v.x);
    atomicAdd(p + 1, v.y);
    atomicAdd(p + 2, v.z);
    atomicAdd(p + 3, v.w);
}

// ------------------------------------------------------------ classifier head

__global__ void head_kernel(const float* __restrict__ pooled, const float* __restrict__ gcnt,
                            const float* __restrict__ Wlin, const float* __restrict__ blin,
                            float* __restrict__ out, int G, int C) {
    int id = blockIdx.x * blockDim.x + threadIdx.x;
    if (id >= G * C) return;
    int g = id / C, c = id % C;
    float inv = 1.0f / fmaxf(gcnt[g], 1.0f);
    float s = 0.0f;
#pragma unroll 8
    for (int h = 0; h < HID; ++h)
        s += pooled[(size_t)g * HID + h] * Wlin[c * HID + h];
    out[id] = s * inv + blin[c];
}

// ---------------------------------------------------------------- launcher

extern "C" void kernel_launch(void* const* d_in, const int* in_sizes, int n_in,
                              void* d_out, int out_size, void* d_ws, size_t ws_size,
                              hipStream_t stream) {
    const int*   x_tokens = (const int*)  d_in[0];
    const int*   edge     = (const int*)  d_in[1];
    const int*   batch    = (const int*)  d_in[2];
    const float* embedW   = (const float*)d_in[3];
    const float* W1l      = (const float*)d_in[4];
    const float* b1l      = (const float*)d_in[5];
    const float* W1r      = (const float*)d_in[6];
    const float* W2l      = (const float*)d_in[7];
    const float* b2l      = (const float*)d_in[8];
    const float* W2r      = (const float*)d_in[9];
    const float* Wlin     = (const float*)d_in[10];
    const float* blin     = (const float*)d_in[11];
    float* out = (float*)d_out;

    const int N = in_sizes[0];          // 50000
    const int E = in_sizes[1] / 2;      // 800000
    const int C = in_sizes[11];         // 5
    const int G = out_size / C;         // 512

    const int* src = edge;
    const int* dst = edge + E;

    // workspace carve-out (all 16B aligned)
    size_t NH = (size_t)N * HID;
    float*    x0     = (float*)d_ws;
    float*    agg    = x0 + NH;
    float*    x1     = agg + NH;
    float*    deg    = x1 + NH;
    float*    pooled = deg + N;
    float*    gcnt   = pooled + (size_t)G * HID;
    unsigned* pw     = (unsigned*)(gcnt + G);
    unsigned* P1l = pw;
    unsigned* P1r = pw + 2048;
    unsigned* P2l = pw + 4096;
    unsigned* P2r = pw + 6144;

    const int TB = 256;
    auto nb = [](long long n, int tb) { return (int)((n + tb - 1) / tb); };

    // zero accumulators (deg, pooled+gcnt contiguous)
    zero_kernel<<<nb(N, TB), TB, 0, stream>>>(deg, (long long)N);
    zero_kernel<<<nb((long long)G * HID + G, TB), TB, 0, stream>>>(pooled, (long long)G * HID + G);

    // degree per dst node, node count per graph
    count_kernel<<<nb(E, TB), TB, 0, stream>>>(dst, deg, E);
    count_kernel<<<nb(N, TB), TB, 0, stream>>>(batch, gcnt, N);

    // pack weights into WMMA B-layout (f16)
    pack_w_kernel<<<8, TB, 0, stream>>>(W1l, P1l);
    pack_w_kernel<<<8, TB, 0, stream>>>(W1r, P1r);
    pack_w_kernel<<<8, TB, 0, stream>>>(W2l, P2l);
    pack_w_kernel<<<8, TB, 0, stream>>>(W2r, P2r);

    // embedding
    embed_kernel<<<nb((long long)N * 16, TB), TB, 0, stream>>>(x_tokens, embedW, x0, N);

    int waves = (N + 15) / 16;
    int sage_blocks = nb((long long)waves * 32, TB);

    // layer 1
    zero_kernel<<<nb(NH, TB), TB, 0, stream>>>(agg, (long long)NH);
    scatter_kernel<<<nb((long long)E * 16, TB), TB, 0, stream>>>(x0, agg, src, dst, E);
    sage_kernel<<<sage_blocks, TB, 0, stream>>>(x0, agg, deg, P1l, P1r, b1l, x1, N);

    // layer 2
    zero_kernel<<<nb(NH, TB), TB, 0, stream>>>(agg, (long long)NH);
    scatter_kernel<<<nb((long long)E * 16, TB), TB, 0, stream>>>(x1, agg, src, dst, E);
    sage_kernel<<<sage_blocks, TB, 0, stream>>>(x1, agg, deg, P2l, P2r, b2l, x0, N);

    // global mean pool + linear head
    pool_kernel<<<nb((long long)N * 16, TB), TB, 0, stream>>>(x0, batch, pooled, N);
    head_kernel<<<nb((long long)G * C, TB), TB, 0, stream>>>(pooled, gcnt, Wlin, blin, out, G, C);
}
